// EISANImodel_83605833384667
// MI455X (gfx1250) — 8-bit matrix operand order repaired and checked by reading, compiled, not run on hardware
//
#include <hip/hip_runtime.h>
#include <cstdint>

static constexpr int B_   = 512;
static constexpr int F_   = 256;
static constexpr int BITS = 8;
static constexpr int E_   = F_ * BITS;
static constexpr int H_   = 8192;
static constexpr int C_   = 100;
static constexpr int THRI = 5;

typedef __attribute__((ext_vector_type(8))) int v8i;
typedef __attribute__((ext_vector_type(4))) int v4i;

__global__ __launch_bounds__(256) void encode_gray(const float* __restrict__ x,
                                                   uint8_t* __restrict__ prev,
                                                   int total  ) {
    int i = blockIdx.x * blockDim.x + threadIdx.x;
    if (i >= total) return;
    float v = x[i];
    v = fminf(fmaxf(v, 0.0f), 1.0f);
    int lv = __float2int_rn(v * 255.0f);
    int g  = lv ^ (lv >> 1);
    uint64_t bytes = 0;
#pragma unroll
    for (int j = 0; j < 8; ++j)
        bytes |= (uint64_t)((g >> j) & 1) << (8 * j);
    *(uint64_t*)(prev + (size_t)i * 8) = bytes;
}

__global__ __launch_bounds__(256) void cvt_f32_i8(const float* __restrict__ w,
                                                  int8_t* __restrict__ o,
                                                  int n4) {
    int i = blockIdx.x * blockDim.x + threadIdx.x;
    if (i >= n4) return;
    float4 v = ((const float4*)w)[i];
    int8_t c0 = (int8_t)__float2int_rn(v.x);
    int8_t c1 = (int8_t)__float2int_rn(v.y);
    int8_t c2 = (int8_t)__float2int_rn(v.z);
    int8_t c3 = (int8_t)__float2int_rn(v.w);
    uint32_t pk = (uint8_t)c0 | ((uint32_t)(uint8_t)c1 << 8) |
                  ((uint32_t)(uint8_t)c2 << 16) | ((uint32_t)(uint8_t)c3 << 24);
    ((uint32_t*)o)[i] = pk;
}

__device__ __forceinline__ v8i load_b_frag(const int8_t* __restrict__ wcol, int k) {
    v4i blo = *(const v4i*)(wcol + k);
    v4i bhi = *(const v4i*)(wcol + k + 32);
    v8i b;
    b[0] = blo[0]; b[1] = blo[1]; b[2] = blo[2]; b[3] = blo[3];
    b[4] = bhi[0]; b[5] = bhi[1]; b[6] = bhi[2]; b[7] = bhi[3];
    return b;
}

__device__ __forceinline__ v8i load_a_frag(const uint8_t* __restrict__ ap) {
    v4i alo = *(const v4i*)(ap);
    v4i ahi = *(const v4i*)(ap + 32);
    v8i a;
    a[0] = alo[0]; a[1] = alo[1]; a[2] = alo[2]; a[3] = alo[3];
    a[4] = ahi[0]; a[5] = ahi[1]; a[6] = ahi[2]; a[7] = ahi[3];
    return a;
}

__global__ __launch_bounds__(256) void gemm_iu8_thresh(
    const uint8_t* __restrict__ Act,
    const int8_t*  __restrict__ W,
    uint8_t*       __restrict__ Out,
    int M, int N, int Ktot) {

    const int lane    = threadIdx.x & 31;
    const int wave    = threadIdx.x >> 5;
    const int laneLo  = lane & 15;
    const int halfSel = lane >> 4;

    const int m0 = blockIdx.y * 64;
    const int n0 = blockIdx.x * 128 + wave * 16;

    const int8_t*  wcol = W + (size_t)(n0 + laneLo) * Ktot + halfSel * 16;
    const uint8_t* ap0  = Act + (size_t)(m0 +  0 + laneLo) * Ktot + halfSel * 16;
    const uint8_t* ap1  = Act + (size_t)(m0 + 16 + laneLo) * Ktot + halfSel * 16;
    const uint8_t* ap2  = Act + (size_t)(m0 + 32 + laneLo) * Ktot + halfSel * 16;
    const uint8_t* ap3  = Act + (size_t)(m0 + 48 + laneLo) * Ktot + halfSel * 16;

    v8i acc[4];
#pragma unroll
    for (int t = 0; t < 4; ++t) acc[t] = (v8i){0,0,0,0,0,0,0,0};

    v8i bfX  = load_b_frag(wcol, 0);
    v8i afX0 = load_a_frag(ap0);
    v8i afX1 = load_a_frag(ap1);
    v8i afX2 = load_a_frag(ap2);
    v8i afX3 = load_a_frag(ap3);

    for (int k = 0; k < Ktot; k += 128) {
        __builtin_prefetch(wcol + k + 128, 0, 0);
        v8i bfY  = load_b_frag(wcol, k + 64);
        v8i afY0 = load_a_frag(ap0 + k + 64);
        v8i afY1 = load_a_frag(ap1 + k + 64);
        v8i afY2 = load_a_frag(ap2 + k + 64);
        v8i afY3 = load_a_frag(ap3 + k + 64);

        acc[0] = __builtin_amdgcn_wmma_i32_16x16x64_iu8(false, afX0, true, bfX, acc[0], false, false);
        acc[1] = __builtin_amdgcn_wmma_i32_16x16x64_iu8(false, afX1, true, bfX, acc[1], false, false);
        acc[2] = __builtin_amdgcn_wmma_i32_16x16x64_iu8(false, afX2, true, bfX, acc[2], false, false);
        acc[3] = __builtin_amdgcn_wmma_i32_16x16x64_iu8(false, afX3, true, bfX, acc[3], false, false);

        if (k + 128 < Ktot) {
            bfX  = load_b_frag(wcol, k + 128);
            afX0 = load_a_frag(ap0 + k + 128);
            afX1 = load_a_frag(ap1 + k + 128);
            afX2 = load_a_frag(ap2 + k + 128);
            afX3 = load_a_frag(ap3 + k + 128);
        }

        acc[0] = __builtin_amdgcn_wmma_i32_16x16x64_iu8(false, afY0, true, bfY, acc[0], false, false);
        acc[1] = __builtin_amdgcn_wmma_i32_16x16x64_iu8(false, afY1, true, bfY, acc[1], false, false);
        acc[2] = __builtin_amdgcn_wmma_i32_16x16x64_iu8(false, afY2, true, bfY, acc[2], false, false);
        acc[3] = __builtin_amdgcn_wmma_i32_16x16x64_iu8(false, afY3, true, bfY, acc[3], false, false);
    }

    const int col = n0 + laneLo;
#pragma unroll
    for (int mt = 0; mt < 4; ++mt) {
#pragma unroll
        for (int g = 0; g < 8; ++g) {
            int m = m0 + mt * 16 + g + halfSel * 8;
            Out[(size_t)m * N + col] = (acc[mt][g] >= THRI) ? (uint8_t)1 : (uint8_t)0;
        }
    }
}

__global__ __launch_bounds__(128) void out_accum(const uint8_t* __restrict__ a0,
                                                 const uint8_t* __restrict__ a1,
                                                 const float* __restrict__ wout,
                                                 float* __restrict__ out,
                                                 int Hh, int C) {
    const int b = blockIdx.x;
    const int c = threadIdx.x;
    if (c >= C) return;
    const float* w0o = wout;
    const float* w1o = wout + (size_t)Hh * C;
    const uint8_t* r0 = a0 + (size_t)b * Hh;
    const uint8_t* r1 = a1 + (size_t)b * Hh;
    float acc = 0.0f;
    for (int h = 0; h < Hh; ++h) {
        float f0 = (float)r0[h];
        float f1 = (float)r1[h];
        acc = fmaf(f0, w0o[(size_t)h * C + c], acc);
        acc = fmaf(f1, w1o[(size_t)h * C + c], acc);
    }
    out[(size_t)b * C + c] = acc;
}

extern "C" void kernel_launch(void* const* d_in, const int* in_sizes, int n_in,
                              void* d_out, int out_size, void* d_ws, size_t ws_size,
                              hipStream_t stream) {
    const float* x    = (const float*)d_in[0];
    const float* w0   = (const float*)d_in[1];
    const float* w1   = (const float*)d_in[2];
    const float* wout = (const float*)d_in[3];
    float* out = (float*)d_out;

    uint8_t* ws   = (uint8_t*)d_ws;
    uint8_t* prev = ws;
    int8_t*  w0i  = (int8_t*)(ws + (size_t)B_ * E_);
    int8_t*  w1i  = (int8_t*)((uint8_t*)w0i + (size_t)H_ * E_);
    uint8_t* a0   = (uint8_t*)w1i + (size_t)H_ * H_;
    uint8_t* a1   = a0 + (size_t)B_ * H_;

    {
        int total = B_ * F_;
        encode_gray<<<(total + 255) / 256, 256, 0, stream>>>(x, prev, total);
    }
    {
        int n4 = (H_ * E_) / 4;
        cvt_f32_i8<<<(n4 + 255) / 256, 256, 0, stream>>>(w0, w0i, n4);
    }
    {
        int n4 = (H_ * H_) / 4;
        cvt_f32_i8<<<(n4 + 255) / 256, 256, 0, stream>>>(w1, w1i, n4);
    }
    {
        dim3 grid(H_ / 128, B_ / 64);
        gemm_iu8_thresh<<<grid, 256, 0, stream>>>(prev, w0i, a0, B_, H_, E_);
    }
    {
        dim3 grid(H_ / 128, B_ / 64);
        gemm_iu8_thresh<<<grid, 256, 0, stream>>>(a0, w1i, a1, B_, H_, H_);
    }
    out_accum<<<B_, 128, 0, stream>>>(a0, a1, wout, out, H_, C_);
}
